// Basic_RSNN_eprop_minsik_88845693485083
// MI455X (gfx1250) — matrix operand order repaired and checked by reading, not compiled, not run on hardware
//
#include <hip/hip_runtime.h>

// ---------------------------------------------------------------------------
// RSNN forward on gfx1250: persistent WMMA kernel.
//  - membrane state lives in accumulator VGPRs for all 250 steps
//  - each of 64 hidden WGs owns a 32-col strip of Wrec^T in 128 KB LDS,
//    stored column-interleaved so both B tiles use immediate DS offsets
//  - each wave computes 2 N-tiles per A fragment (halves L2 spike traffic)
//  - spikes ping-pong through an L2-resident buffer, 1 global barrier/step
// ---------------------------------------------------------------------------

typedef __attribute__((ext_vector_type(16))) _Float16 v16h;
typedef __attribute__((ext_vector_type(8)))  float    v8f;

#define TAU   0.6f
#define THR   0.6f
#define BATCH 128
#define TSTEPS 250
#define NIN   700
#define NINP  704      // K padded to multiple of 32
#define NHID  2048
#define NOUT  20
#define NOUTP 32       // N padded to 2 WMMA tiles
#define NWG_HID 64     // 32-col strips
#define NWG_TOT (NWG_HID + 1)
#define LDS_BYTES (32 * NHID * 2)   // 128 KB weight strip

// workspace layout (bytes), all 16B aligned
#define OFF_CNT   0u                                        // 256 * 4 = 1 KB
#define OFF_HS    4096u                                     // 2*128*2048 f16 = 1 MB
#define OFF_W1H   (OFF_HS + 2u * BATCH * NHID * 2u)         // 2048*704 f16
#define OFF_WRECT (OFF_W1H + (unsigned)NHID * NINP * 2u)    // 2048*2048 f16 (transposed)
#define OFF_WOUT  (OFF_WRECT + (unsigned)NHID * NHID * 2u)  // 32*2048 f16
#define OFF_XH    (OFF_WOUT + (unsigned)NOUTP * NHID * 2u)  // 250*128*704 f16 (~45 MB)

union FragH {
  v16h  h;
  uint4 q[2];
};

static __device__ inline v8f wmma_f16(v16h a, v16h b, v8f c) {
  return __builtin_amdgcn_wmma_f32_16x16x32_f16(false, a, false, b, (short)0, c,
                                                false, false);
}

static __device__ inline v8f zero8() {
  v8f z;
#pragma unroll
  for (int i = 0; i < 8; ++i) z[i] = 0.f;
  return z;
}

// ------------------------------- prep kernels ------------------------------

__global__ void k_zero_u32(unsigned* __restrict__ p, unsigned n) {
  unsigned i = blockIdx.x * blockDim.x + threadIdx.x;
  if (i < n) p[i] = 0u;
}

__global__ void k_cvt_x(const float* __restrict__ x, _Float16* __restrict__ xh) {
  size_t i = (size_t)blockIdx.x * blockDim.x + threadIdx.x;
  if (i >= (size_t)TSTEPS * BATCH * NINP) return;
  unsigned k = (unsigned)(i % NINP);
  size_t bt  = i / NINP;
  unsigned b = (unsigned)(bt % BATCH);
  unsigned t = (unsigned)(bt / BATCH);
  float v = (k < NIN) ? x[((size_t)b * TSTEPS + t) * NIN + k] : 0.f;
  xh[i] = (_Float16)v;   // xh: [t][b][kpad]
}

__global__ void k_cvt_w1(const float* __restrict__ w, _Float16* __restrict__ wh) {
  size_t i = (size_t)blockIdx.x * blockDim.x + threadIdx.x;
  if (i >= (size_t)NHID * NINP) return;
  unsigned k = (unsigned)(i % NINP);
  size_t   n = i / NINP;
  wh[i] = (_Float16)((k < NIN) ? w[n * NIN + k] : 0.f);   // [n][kpad]
}

__global__ void k_cvt_wrect(const float* __restrict__ w, _Float16* __restrict__ wt) {
  size_t i = (size_t)blockIdx.x * blockDim.x + threadIdx.x;
  if (i >= (size_t)NHID * NHID) return;
  size_t   n = i / NHID;
  unsigned k = (unsigned)(i % NHID);
  wt[i] = (_Float16)w[(size_t)k * NHID + n];   // wt[n][k] = Wrec[k][n]
}

__global__ void k_cvt_wout(const float* __restrict__ w, _Float16* __restrict__ wh) {
  size_t i = (size_t)blockIdx.x * blockDim.x + threadIdx.x;
  if (i >= (size_t)NOUTP * NHID) return;
  size_t   n = i / NHID;
  unsigned k = (unsigned)(i % NHID);
  wh[i] = (_Float16)((n < NOUT) ? w[n * NHID + k] : 0.f);  // [npad][k]
}

// ------------------------------- main kernel -------------------------------

__global__ __launch_bounds__(256) void k_rsnn(
    const _Float16* __restrict__ xh,     // [250][128][704]
    const _Float16* __restrict__ w1h,    // [2048][704]
    const _Float16* __restrict__ wrect,  // [2048][2048]  ([n][k])
    const _Float16* __restrict__ wouth,  // [32][2048]    ([n][k])
    _Float16* __restrict__ hsbuf,        // [2][128][2048] spike ping-pong
    int* __restrict__ cnt,               // [>=250] arrival counters (pre-zeroed)
    float* __restrict__ out)             // [128][250][20]
{
  extern __shared__ _Float16 ldsW[];     // 32 cols x 2048 K = 128 KB

  const int g    = blockIdx.x;
  const int tid  = threadIdx.x;
  const int lane = tid & 31;
  const int wave = tid >> 5;
  const bool isHid = (g < NWG_HID);

  // Stage this WG's constant 32-col weight strip into LDS once.
  // Interleave: source col n -> LDS row ((n&15)*2 + (n>>4)) so the pair of
  // B columns used by one wave is only 4 KB apart (immediate DS offsets).
  {
    const _Float16* src = isHid ? (wrect + (size_t)g * 32 * NHID) : wouth;
    const uint4* s4 = (const uint4*)src;
    uint4* d4 = (uint4*)ldsW;
#pragma unroll 4
    for (int i = tid; i < (32 * NHID) / 8; i += 256) {
      int n   = i >> 8;          // 2048/8 = 256 uint4 per column
      int off = i & 255;
      int dn  = ((n & 15) << 1) | (n >> 4);
      d4[dn * 256 + off] = s4[i];
    }
  }
  __syncthreads();

  // WMMA fragment addressing (contiguous K: lane half h holds k 16h..16h+15 in A and B)
  const int mbase = wave * 16;                 // this wave's 16 batch rows
  const int rowA  = mbase + (lane & 15);       // A: row per lane
  const int koffA = (lane >> 4) << 4;          // A: lanes>=16 hold K 16..31
  const int koffB = (lane >> 4) << 4;          // B: lanes>=16 hold K 16..31
  const int nsub  = lane & 15;                 // B/C: column within 16-wide tile
  const int rowC  = mbase + ((lane >> 4) << 3);// C: VGPR j -> row rowC+j

  v8f mem0 = zero8(), mem1 = zero8();   // membrane tiles (register-resident)
  v8f spk0 = zero8(), spk1 = zero8();   // previous spikes for the same tiles

  const _Float16* lb0 = ldsW + (size_t)(nsub * 2) * NHID;  // B col nsub
  const _Float16* lb1 = lb0 + NHID;                        // B col nsub+16 (+4KB)

  for (int t = 0; t < TSTEPS; ++t) {
    const _Float16* hs_r = hsbuf + (size_t)(t & 1) * BATCH * NHID;        // prev
    _Float16*       hs_w = hsbuf + (size_t)((t & 1) ^ 1) * BATCH * NHID;  // new

    if (isHid) {
      v8f acc0 = zero8(), acc1 = zero8();
      // ---- input current: xt @ W1^T  (K = 704, B strip from L2) ----
      const _Float16* xa  = xh + ((size_t)t * BATCH + rowA) * NINP;
      const _Float16* wa0 = w1h + (size_t)(g * 32 + nsub) * NINP;
      const _Float16* wa1 = wa0 + (size_t)16 * NINP;
#pragma unroll 2
      for (int kb = 0; kb < NINP; kb += 32) {
        FragH a, b0, b1;
        const uint4* pa = (const uint4*)(xa + kb + koffA);
        a.q[0] = pa[0]; a.q[1] = pa[1];
        const uint4* p0 = (const uint4*)(wa0 + kb + koffB);
        b0.q[0] = p0[0]; b0.q[1] = p0[1];
        const uint4* p1 = (const uint4*)(wa1 + kb + koffB);
        b1.q[0] = p1[0]; b1.q[1] = p1[1];
        acc0 = wmma_f16(a.h, b0.h, acc0);
        acc1 = wmma_f16(a.h, b1.h, acc1);
      }
      // ---- recurrent current: hs @ Wrec  (K = 2048, B from LDS) ----
      const _Float16* ha = hs_r + (size_t)rowA * NHID;
#pragma unroll 4
      for (int kb = 0; kb < NHID; kb += 32) {
        FragH a, b0, b1;
        const uint4* pa = (const uint4*)(ha + kb + koffA);
        a.q[0] = pa[0]; a.q[1] = pa[1];
        const uint4* p0 = (const uint4*)(lb0 + kb + koffB);
        b0.q[0] = p0[0]; b0.q[1] = p0[1];
        const uint4* p1 = (const uint4*)(lb1 + kb + koffB);
        b1.q[0] = p1[0]; b1.q[1] = p1[1];
        acc0 = wmma_f16(a.h, b0.h, acc0);
        acc1 = wmma_f16(a.h, b1.h, acc1);
      }
      // ---- LIF update (hard reset) + spike emit ----
      const int c0 = g * 32 + nsub;
#pragma unroll
      for (int j = 0; j < 8; ++j) {
        float m0 = TAU * mem0[j] * (1.f - spk0[j]) + acc0[j];
        float s0 = (m0 > THR) ? 1.f : 0.f;
        mem0[j] = m0; spk0[j] = s0;
        float m1 = TAU * mem1[j] * (1.f - spk1[j]) + acc1[j];
        float s1 = (m1 > THR) ? 1.f : 0.f;
        mem1[j] = m1; spk1[j] = s1;
        _Float16* hrow = hs_w + (size_t)(rowC + j) * NHID + c0;
        hrow[0]  = (_Float16)s0;
        hrow[16] = (_Float16)s1;
      }
    }

    // ---- device-wide barrier: all spikes of step t visible ----
    __threadfence();
    __syncthreads();
    if (tid == 0) {
      __hip_atomic_fetch_add(&cnt[t], 1, __ATOMIC_RELEASE, __HIP_MEMORY_SCOPE_AGENT);
      while (__hip_atomic_load(&cnt[t], __ATOMIC_ACQUIRE, __HIP_MEMORY_SCOPE_AGENT)
             < NWG_TOT)
        __builtin_amdgcn_s_sleep(2);
    }
    __syncthreads();

    if (!isHid) {
      // ---- output layer: hs_new @ Wout^T  (K = 2048, B from LDS) ----
      v8f acc0 = zero8(), acc1 = zero8();
      const _Float16* ha = hs_w + (size_t)rowA * NHID;
#pragma unroll 4
      for (int kb = 0; kb < NHID; kb += 32) {
        FragH a, b0, b1;
        const uint4* pa = (const uint4*)(ha + kb + koffA);
        a.q[0] = pa[0]; a.q[1] = pa[1];
        const uint4* p0 = (const uint4*)(lb0 + kb + koffB);
        b0.q[0] = p0[0]; b0.q[1] = p0[1];
        const uint4* p1 = (const uint4*)(lb1 + kb + koffB);
        b1.q[0] = p1[0]; b1.q[1] = p1[1];
        acc0 = wmma_f16(a.h, b0.h, acc0);
        acc1 = wmma_f16(a.h, b1.h, acc1);
      }
#pragma unroll
      for (int j = 0; j < 8; ++j) {
        float m0 = TAU * mem0[j] * (1.f - spk0[j]) + acc0[j];
        float s0 = (m0 > THR) ? 1.f : 0.f;
        mem0[j] = m0; spk0[j] = s0;
        float m1 = TAU * mem1[j] * (1.f - spk1[j]) + acc1[j];
        float s1 = (m1 > THR) ? 1.f : 0.f;
        mem1[j] = m1; spk1[j] = s1;
        float* orow = out + ((size_t)(rowC + j) * TSTEPS + t) * NOUT;
        orow[nsub] = s0;                       // cols 0..15 all valid (<20)
        if (nsub < NOUT - 16) orow[nsub + 16] = s1;  // cols 16..19
      }
    }
  }
}

// ------------------------------- host launch -------------------------------

extern "C" void kernel_launch(void* const* d_in, const int* in_sizes, int n_in,
                              void* d_out, int out_size, void* d_ws, size_t ws_size,
                              hipStream_t stream) {
  (void)in_sizes; (void)n_in; (void)out_size; (void)ws_size;

  const float* x    = (const float*)d_in[0];   // [128,250,700]
  const float* W1   = (const float*)d_in[1];   // [2048,700]
  const float* Wrec = (const float*)d_in[2];   // [2048,2048]
  const float* Wout = (const float*)d_in[3];   // [20,2048]
  float* out = (float*)d_out;                  // [128,250,20]
  char*  ws  = (char*)d_ws;

  int*      cnt   = (int*)(ws + OFF_CNT);
  _Float16* hsbuf = (_Float16*)(ws + OFF_HS);
  _Float16* w1h   = (_Float16*)(ws + OFF_W1H);
  _Float16* wrect = (_Float16*)(ws + OFF_WRECT);
  _Float16* wouth = (_Float16*)(ws + OFF_WOUT);
  _Float16* xh    = (_Float16*)(ws + OFF_XH);

  // allow 128 KB dynamic LDS (attribute set is not a stream op; capture-safe)
  static_cast<void>(hipFuncSetAttribute(
      reinterpret_cast<const void*>(&k_rsnn),
      hipFuncAttributeMaxDynamicSharedMemorySize, LDS_BYTES));

  // zero barrier counters + both spike buffers (contiguous from ws base)
  {
    unsigned n = (OFF_HS + 2u * BATCH * NHID * 2u) / 4u;
    k_zero_u32<<<(n + 255) / 256, 256, 0, stream>>>((unsigned*)ws, n);
  }
  {
    size_t n = (size_t)NHID * NINP;
    k_cvt_w1<<<(unsigned)((n + 255) / 256), 256, 0, stream>>>(W1, w1h);
  }
  {
    size_t n = (size_t)NHID * NHID;
    k_cvt_wrect<<<(unsigned)((n + 255) / 256), 256, 0, stream>>>(Wrec, wrect);
  }
  {
    size_t n = (size_t)NOUTP * NHID;
    k_cvt_wout<<<(unsigned)((n + 255) / 256), 256, 0, stream>>>(Wout, wouth);
  }
  {
    size_t n = (size_t)TSTEPS * BATCH * NINP;
    k_cvt_x<<<(unsigned)((n + 255) / 256), 256, 0, stream>>>(x, xh);
  }

  k_rsnn<<<NWG_TOT, 256, LDS_BYTES, stream>>>(xh, w1h, wrect, wouth, hsbuf,
                                              cnt, out);
}
